// FE_12343736008796
// MI455X (gfx1250) — compile-verified, matrix operand order checked by reading, not run on hardware
//
#include <hip/hip_runtime.h>
#include <hip/hip_bf16.h>
#include <math.h>

// ---------------------------------------------------------------------------
// FE block for MI455X (gfx1250): f16 WMMA GEMMs + VALU attention/conv/LN.
// ---------------------------------------------------------------------------

typedef _Float16 f16;
typedef f16   v16h __attribute__((ext_vector_type(16)));
typedef f16   v8h  __attribute__((ext_vector_type(8)));
typedef float v8f  __attribute__((ext_vector_type(8)));

#define B_  8
#define H_  56
#define W_  56
#define C_  512
#define C2_ 256
#define NH_ 8
#define D_  32
#define HW_ (H_ * W_)          // 3136
#define T_  (B_ * HW_)         // 25088
#define Q_  49

// ---------------------------------------------------------------------------
// Tile loader: 16x32 f16 tile in WMMA A/B register layout.
// lanes 0-15 : row r0+lane,   halves[0..7]=K+0..7,  halves[8..15]=K+16..23
// lanes 16-31: row r0+lane-16, halves[0..7]=K+8..15, halves[8..15]=K+24..31
// ---------------------------------------------------------------------------
__device__ __forceinline__ v16h load_tile16x32(const f16* __restrict__ base,
                                               int ld, int r0, int k0,
                                               int lane, int rmax) {
  int r = r0 + (lane & 15);
  if (r >= rmax) r = rmax - 1;          // clamp (stores are guarded)
  const int kb = (lane < 16) ? 0 : 8;
  const f16* p = base + (size_t)r * ld + k0 + kb;
  v8h lo = *(const v8h*)(p);
  v8h hi = *(const v8h*)(p + 16);
  v16h out;
#pragma unroll
  for (int i = 0; i < 8; ++i) { out[i] = lo[i]; out[i + 8] = hi[i]; }
  return out;
}

__device__ __forceinline__ float gelu_exact(float x) {
  return 0.5f * x * (1.0f + erff(x * 0.70710678118654752f));
}

// ---------------------------------------------------------------------------
// Generic WMMA GEMM:  out = epilogue(A[M,K] @ BT[N,K]^T + bias)
// 256 threads = 8 waves; each wave -> 32x32 tile; block -> 128(M) x 64(N).
// MODE 0: f16 out              MODE 1: gelu -> f16 out
// MODE 2: (* aux) -> f16 out   MODE 3: f32 out (ld = N)
// MODE 4: split f32 (col<512 -> outF0, else outF1; ld = 512)
// ---------------------------------------------------------------------------
template <int MODE>
__global__ __launch_bounds__(256) void gemm_wmma_kernel(
    const f16* __restrict__ A, const f16* __restrict__ BT,
    const float* __restrict__ bias, int M, int N, int K,
    f16* __restrict__ outH, int ldo, int colOff,
    const f16* __restrict__ aux, int ldaux,
    float* __restrict__ outF0, float* __restrict__ outF1) {
  const int lane = threadIdx.x & 31;
  const int wave = threadIdx.x >> 5;
  const int wm = wave & 3, wn = wave >> 2;
  const int m0 = blockIdx.x * 128 + wm * 32;
  const int n0 = blockIdx.y * 64 + wn * 32;
  if (m0 >= M) return;                   // wave-uniform exit (EXEC stays all-1)

  v8f a00 = {}, a01 = {}, a10 = {}, a11 = {};
  for (int kt = 0; kt < K; kt += 32) {
    v16h ta0 = load_tile16x32(A, K, m0, kt, lane, M);
    v16h ta1 = load_tile16x32(A, K, m0 + 16, kt, lane, M);
    v16h tb0 = load_tile16x32(BT, K, n0, kt, lane, N);
    v16h tb1 = load_tile16x32(BT, K, n0 + 16, kt, lane, N);
    a00 = __builtin_amdgcn_wmma_f32_16x16x32_f16(false, ta0, false, tb0, (short)0, a00, false, false);
    a01 = __builtin_amdgcn_wmma_f32_16x16x32_f16(false, ta0, false, tb1, (short)0, a01, false, false);
    a10 = __builtin_amdgcn_wmma_f32_16x16x32_f16(false, ta1, false, tb0, (short)0, a10, false, false);
    a11 = __builtin_amdgcn_wmma_f32_16x16x32_f16(false, ta1, false, tb1, (short)0, a11, false, false);
  }

  v8f accs[2][2] = {{a00, a01}, {a10, a11}};
  const int colBase = lane & 15;
  const int rAdd = (lane < 16) ? 0 : 8;
#pragma unroll
  for (int mi = 0; mi < 2; ++mi) {
#pragma unroll
    for (int ni = 0; ni < 2; ++ni) {
      const int col = n0 + ni * 16 + colBase;
      const int rb = m0 + mi * 16 + rAdd;
      const float bv = bias ? bias[col] : 0.0f;
#pragma unroll
      for (int v = 0; v < 8; ++v) {
        const int r = rb + v;
        if (r >= M) continue;
        float xv = accs[mi][ni][v] + bv;
        if (MODE == 1) xv = gelu_exact(xv);
        if (MODE == 2) xv *= (float)aux[(size_t)r * ldaux + col];
        if (MODE <= 2) {
          outH[(size_t)r * ldo + colOff + col] = (f16)xv;
        } else if (MODE == 3) {
          outF0[(size_t)r * N + col] = xv;
        } else {  // MODE 4
          if (col < 512) outF0[(size_t)r * 512 + col] = xv;
          else           outF1[(size_t)r * 512 + (col - 512)] = xv;
        }
      }
    }
  }
}

// ---------------------------------------------------------------------------
// Weight transpose + f32->f16:  W[K,N] row-major -> WT[N,K] row-major
// ---------------------------------------------------------------------------
__global__ __launch_bounds__(256) void transpose_w_kernel(
    const float* __restrict__ W, f16* __restrict__ WT, int K, int N) {
  int idx = blockIdx.x * 256 + threadIdx.x;
  if (idx >= K * N) return;
  int k = idx / N, n = idx % N;
  WT[(size_t)n * K + k] = (f16)W[idx];
}

__global__ __launch_bounds__(256) void concat_bias_kernel(
    const float* __restrict__ a, const float* __restrict__ b,
    float* __restrict__ o, int n) {
  int i = blockIdx.x * 256 + threadIdx.x;
  if (i < n) o[i] = a[i];
  else if (i < 2 * n) o[i] = b[i - n];
}

// ---------------------------------------------------------------------------
// LayerNorm over C=512, one token per block (128 threads, 4 ch/thread) -> f16
// ---------------------------------------------------------------------------
__global__ __launch_bounds__(128) void ln_kernel(
    const float* __restrict__ x, const float* __restrict__ w,
    const float* __restrict__ b, f16* __restrict__ out) {
  __shared__ float red[128];
  const int t = blockIdx.x, tid = threadIdx.x;
  const float* px = x + (size_t)t * C_;
  float v[4];
  float s = 0.0f;
#pragma unroll
  for (int i = 0; i < 4; ++i) { v[i] = px[tid + i * 128]; s += v[i]; }
  red[tid] = s; __syncthreads();
  for (int o = 64; o > 0; o >>= 1) { if (tid < o) red[tid] += red[tid + o]; __syncthreads(); }
  const float mu = red[0] * (1.0f / C_);
  __syncthreads();
  s = 0.0f;
#pragma unroll
  for (int i = 0; i < 4; ++i) { float d = v[i] - mu; s += d * d; }
  red[tid] = s; __syncthreads();
  for (int o = 64; o > 0; o >>= 1) { if (tid < o) red[tid] += red[tid + o]; __syncthreads(); }
  const float rs = rsqrtf(red[0] * (1.0f / C_) + 1e-6f);
#pragma unroll
  for (int i = 0; i < 4; ++i) {
    int c = tid + i * 128;
    out[(size_t)t * C_ + c] = (f16)((v[i] - mu) * rs * w[c] + b[c]);
  }
}

// ---------------------------------------------------------------------------
// 8x8 mean pool of concat(xn, xen) -> pooled f16 [B*49, 1024]
// ---------------------------------------------------------------------------
__global__ __launch_bounds__(256) void pool_kernel(
    const f16* __restrict__ xn, const f16* __restrict__ xen,
    f16* __restrict__ pooled) {
  const int bq = blockIdx.x;
  const int b = bq / Q_, q = bq % Q_;
  const int wy = q / 7, wx = q % 7;
  const int tid = threadIdx.x;
#pragma unroll
  for (int i = 0; i < 4; ++i) {
    const int ch = tid + i * 256;
    const f16* src = (ch < C_) ? xn : xen;
    const int c = ch & (C_ - 1);
    float s = 0.0f;
    for (int iy = 0; iy < 8; ++iy) {
      const int y = wy * 8 + iy;
      for (int ix = 0; ix < 8; ++ix) {
        const int x = wx * 8 + ix;
        s += (float)src[(((size_t)b * H_ + y) * W_ + x) * C_ + c];
      }
    }
    pooled[(size_t)bq * 1024 + ch] = (f16)(s * (1.0f / 64.0f));
  }
}

// ---------------------------------------------------------------------------
// Pooled-query attention: one block per (b, head, q) row.
// scores[3136] live in LDS; softmax in f32; output reduced via ds_add_f32.
// ---------------------------------------------------------------------------
__global__ __launch_bounds__(256) void attn_kernel(
    const float* __restrict__ m_f, const f16* __restrict__ kv,
    float* __restrict__ attn_o) {
  const int idx = blockIdx.x;
  const int q = idx % Q_;
  const int bh = idx / Q_;
  const int h = bh % NH_, b = bh / NH_;
  __shared__ float sc[HW_];
  __shared__ float red[256];
  __shared__ float qv[D_];
  __shared__ float oacc[D_];
  const int tid = threadIdx.x;
  if (tid < D_) {
    qv[tid] = m_f[(size_t)(b * Q_ + q) * C2_ + h * D_ + tid] * 0.17677669529663687f;
    oacc[tid] = 0.0f;
  }
  __syncthreads();

  const f16* kbase = kv + (size_t)b * HW_ * C_ + h * D_;
  float lmax = -1e30f;
  for (int key = tid; key < HW_; key += 256) {
    const f16* kp = kbase + (size_t)key * C_;
    float s = 0.0f;
#pragma unroll
    for (int d = 0; d < D_; ++d) s += qv[d] * (float)kp[d];
    sc[key] = s;
    lmax = fmaxf(lmax, s);
  }
  red[tid] = lmax; __syncthreads();
  for (int o = 128; o > 0; o >>= 1) { if (tid < o) red[tid] = fmaxf(red[tid], red[tid + o]); __syncthreads(); }
  const float mx = red[0];
  __syncthreads();

  float lsum = 0.0f;
  float acc[D_];
#pragma unroll
  for (int d = 0; d < D_; ++d) acc[d] = 0.0f;
  const f16* vbase = kbase + C2_;  // v occupies cols 256..511 of kv
  for (int key = tid; key < HW_; key += 256) {
    const float p = __expf(sc[key] - mx);
    lsum += p;
    const f16* vp = vbase + (size_t)key * C_;
#pragma unroll
    for (int d = 0; d < D_; ++d) acc[d] += p * (float)vp[d];
  }
  red[tid] = lsum; __syncthreads();
  for (int o = 128; o > 0; o >>= 1) { if (tid < o) red[tid] += red[tid + o]; __syncthreads(); }
  const float tot = red[0];
#pragma unroll
  for (int d = 0; d < D_; ++d) atomicAdd(&oacc[d], acc[d]);
  __syncthreads();
  if (tid < D_) attn_o[((size_t)bh * Q_ + q) * D_ + tid] = oacc[tid] / tot;
}

// ---------------------------------------------------------------------------
// Depthwise 7x7 conv (pad 3), channels-last, f32 accumulate.
// ---------------------------------------------------------------------------
__global__ __launch_bounds__(256) void dwconv_kernel(
    const f16* __restrict__ e1, const float* __restrict__ cw,
    const float* __restrict__ cb, f16* __restrict__ out) {
  const int pix = blockIdx.x;
  const int b = pix / HW_;
  const int yx = pix % HW_;
  const int y = yx / W_, x = yx % W_;
  const int c = threadIdx.x;
  float s = cb[c];
  for (int dy = 0; dy < 7; ++dy) {
    const int yy = y + dy - 3;
    if (yy < 0 || yy >= H_) continue;
    for (int dx = 0; dx < 7; ++dx) {
      const int xx = x + dx - 3;
      if (xx < 0 || xx >= W_) continue;
      s += (float)e1[(((size_t)b * H_ + yy) * W_ + xx) * C2_ + c] *
           cw[c * 49 + dy * 7 + dx];
    }
  }
  out[(size_t)pix * C2_ + c] = (f16)s;
}

// ---------------------------------------------------------------------------
// Bilinear 7x7 -> 56x56 upsample (half-pixel centers), writes xcat cols 0..255
// ---------------------------------------------------------------------------
__global__ __launch_bounds__(256) void upsample_kernel(
    const float* __restrict__ attn_small, f16* __restrict__ xcat) {
  const int pix = blockIdx.x;
  const int b = pix / HW_;
  const int yx = pix % HW_;
  const int y = yx / W_, x = yx % W_;
  const float sy = (y + 0.5f) * 0.125f - 0.5f;
  const float sx = (x + 0.5f) * 0.125f - 0.5f;
  const int y0 = (int)floorf(sy), x0 = (int)floorf(sx);
  const float fy = sy - y0, fx = sx - x0;
  const int y0c = min(6, max(0, y0)),     x0c = min(6, max(0, x0));
  const int y1c = min(6, max(0, y0 + 1)), x1c = min(6, max(0, x0 + 1));
  const int c = threadIdx.x;
  const int head = c >> 5, dd = c & 31;
  const float* base = attn_small + ((size_t)(b * NH_ + head) * Q_) * D_ + dd;
  const float g00 = base[(y0c * 7 + x0c) * D_];
  const float g01 = base[(y0c * 7 + x1c) * D_];
  const float g10 = base[(y1c * 7 + x0c) * D_];
  const float g11 = base[(y1c * 7 + x1c) * D_];
  const float v = (1.0f - fy) * ((1.0f - fx) * g00 + fx * g01) +
                  fy * ((1.0f - fx) * g10 + fx * g11);
  xcat[(size_t)pix * C_ + c] = (f16)v;
}

// ---------------------------------------------------------------------------
// Host launcher
// ---------------------------------------------------------------------------
static inline size_t align256(size_t x) { return (x + 255) & ~(size_t)255; }

extern "C" void kernel_launch(void* const* d_in, const int* in_sizes, int n_in,
                              void* d_out, int out_size, void* d_ws, size_t ws_size,
                              hipStream_t stream) {
  (void)in_sizes; (void)n_in; (void)out_size; (void)ws_size;

  const float* x       = (const float*)d_in[0];
  const float* x_e     = (const float*)d_in[1];
  const float* norm_w  = (const float*)d_in[2];
  const float* norm_b  = (const float*)d_in[3];
  const float* norme_w = (const float*)d_in[4];
  const float* norme_b = (const float*)d_in[5];
  const float* qcut_w  = (const float*)d_in[6];
  const float* qcut_b  = (const float*)d_in[7];
  const float* l_w     = (const float*)d_in[8];
  const float* l_b     = (const float*)d_in[9];
  const float* kv_w    = (const float*)d_in[10];
  const float* kv_b    = (const float*)d_in[11];
  const float* sc_w    = (const float*)d_in[12];
  const float* sc_b    = (const float*)d_in[13];
  const float* efore_w = (const float*)d_in[14];
  const float* efore_b = (const float*)d_in[15];
  const float* econv_w = (const float*)d_in[16];
  const float* econv_b = (const float*)d_in[17];
  const float* eback_w = (const float*)d_in[18];
  const float* eback_b = (const float*)d_in[19];
  const float* proj_w  = (const float*)d_in[20];
  const float* proj_b  = (const float*)d_in[21];
  const float* proje_w = (const float*)d_in[22];
  const float* proje_b = (const float*)d_in[23];

  float* x_out   = (float*)d_out;                       // [T, 512]
  float* x_e_out = (float*)d_out + (size_t)T_ * C_;     // [T, 512]

  // workspace carve-up
  char* ws = (char*)d_ws;
  size_t off = 0;
  auto carve = [&](size_t bytes) { char* p = ws + off; off += align256(bytes); return p; };

  f16* xn_h    = (f16*)carve((size_t)T_ * C_ * 2);
  f16* xen_h   = (f16*)carve((size_t)T_ * C_ * 2);
  f16* kv_h    = (f16*)carve((size_t)T_ * C_ * 2);
  f16* bx_h    = (f16*)carve((size_t)T_ * C_ * 2);   // bfeat, later reused as xcat
  f16* cut_h   = (f16*)carve((size_t)T_ * C2_ * 2);
  f16* e1_h    = (f16*)carve((size_t)T_ * C2_ * 2);
  f16* ec_h    = (f16*)carve((size_t)T_ * C2_ * 2);
  f16* pool_h  = (f16*)carve((size_t)B_ * Q_ * 1024 * 2);
  float* m_f   = (float*)carve((size_t)B_ * Q_ * C2_ * 4);
  float* attn_s= (float*)carve((size_t)B_ * NH_ * Q_ * D_ * 4);
  f16* qcutT   = (f16*)carve((size_t)C2_ * C_ * 2);
  f16* lT      = (f16*)carve((size_t)C_ * C_ * 2);
  f16* kvT     = (f16*)carve((size_t)C_ * C_ * 2);
  f16* scT     = (f16*)carve((size_t)C2_ * 1024 * 2);
  f16* eforeT  = (f16*)carve((size_t)C2_ * C_ * 2);
  f16* ebackT  = (f16*)carve((size_t)C2_ * C2_ * 2);
  f16* projT   = (f16*)carve((size_t)1024 * C_ * 2);  // [proj^T ; proje^T]
  float* bias2 = (float*)carve(1024 * 4);

  auto tgrid = [](int n) { return dim3((n + 255) / 256); };

  // 1) weight transposes (f32 [K,N] -> f16 [N,K])
  transpose_w_kernel<<<tgrid(C_ * C2_), 256, 0, stream>>>(qcut_w, qcutT, C_, C2_);
  transpose_w_kernel<<<tgrid(C_ * C_), 256, 0, stream>>>(l_w, lT, C_, C_);
  transpose_w_kernel<<<tgrid(C_ * C_), 256, 0, stream>>>(kv_w, kvT, C_, C_);
  transpose_w_kernel<<<tgrid(1024 * C2_), 256, 0, stream>>>(sc_w, scT, 1024, C2_);
  transpose_w_kernel<<<tgrid(C_ * C2_), 256, 0, stream>>>(efore_w, eforeT, C_, C2_);
  transpose_w_kernel<<<tgrid(C2_ * C2_), 256, 0, stream>>>(eback_w, ebackT, C2_, C2_);
  transpose_w_kernel<<<tgrid(C_ * C_), 256, 0, stream>>>(proj_w, projT, C_, C_);
  transpose_w_kernel<<<tgrid(C_ * C_), 256, 0, stream>>>(proje_w, projT + (size_t)C_ * C_, C_, C_);
  concat_bias_kernel<<<4, 256, 0, stream>>>(proj_b, proje_b, bias2, C_);

  // 2) LayerNorms -> f16
  ln_kernel<<<T_, 128, 0, stream>>>(x, norm_w, norm_b, xn_h);
  ln_kernel<<<T_, 128, 0, stream>>>(x_e, norme_w, norme_b, xen_h);

  // 3) 8x8 mean pool -> pooled [B*49, 1024]
  pool_kernel<<<B_ * Q_, 256, 0, stream>>>(xn_h, xen_h, pool_h);

  // 4) m = pooled @ sc_w + sc_b  (M=392, K=1024, N=256) -> f32
  gemm_wmma_kernel<3><<<dim3(4, 4), 256, 0, stream>>>(
      pool_h, scT, sc_b, B_ * Q_, C2_, 1024, nullptr, 0, 0, nullptr, 0, m_f, nullptr);

  // 5) cutted = xn @ qcut_w  (25088 x 512 -> 256)
  gemm_wmma_kernel<0><<<dim3(T_ / 128, 4), 256, 0, stream>>>(
      xn_h, qcutT, qcut_b, T_, C2_, C_, cut_h, C2_, 0, nullptr, 0, nullptr, nullptr);

  // 6) bfeat = gelu(xn @ l_w)  (25088 x 512 -> 512)
  gemm_wmma_kernel<1><<<dim3(T_ / 128, 8), 256, 0, stream>>>(
      xn_h, lT, l_b, T_, C_, C_, bx_h, C_, 0, nullptr, 0, nullptr, nullptr);

  // 7) kv = bfeat @ kv_w  (k = cols 0..255, v = cols 256..511)
  gemm_wmma_kernel<0><<<dim3(T_ / 128, 8), 256, 0, stream>>>(
      bx_h, kvT, kv_b, T_, C_, C_, kv_h, C_, 0, nullptr, 0, nullptr, nullptr);

  // 8) attention (softmax over 3136 keys, per (b,h,q))
  attn_kernel<<<B_ * NH_ * Q_, 256, 0, stream>>>(m_f, kv_h, attn_s);

  // 9) e1 = xen @ efore_w
  gemm_wmma_kernel<0><<<dim3(T_ / 128, 4), 256, 0, stream>>>(
      xen_h, eforeT, efore_b, T_, C2_, C_, e1_h, C2_, 0, nullptr, 0, nullptr, nullptr);

  // 10) depthwise 7x7 conv
  dwconv_kernel<<<T_, 256, 0, stream>>>(e1_h, econv_w, econv_b, ec_h);

  // 11) xcat[:,256:512] = (conv @ eback_w + b) * cutted    (bx_h now = xcat)
  gemm_wmma_kernel<2><<<dim3(T_ / 128, 4), 256, 0, stream>>>(
      ec_h, ebackT, eback_b, T_, C2_, C2_, bx_h, C_, C2_, cut_h, C2_, nullptr, nullptr);

  // 12) xcat[:,0:256] = bilinear upsample of attention
  upsample_kernel<<<T_, 256, 0, stream>>>(attn_s, bx_h);

  // 13) x_out / x_e_out = xcat @ [proj_w | proje_w]  (N = 1024, split f32)
  gemm_wmma_kernel<4><<<dim3(T_ / 128, 16), 256, 0, stream>>>(
      bx_h, projT, bias2, T_, 1024, C_, nullptr, 0, 0, nullptr, 0, x_out, x_e_out);
}
